// KoopmanMlpExtractor_50663434223735
// MI455X (gfx1250) — matrix operand order repaired and checked by reading, compiled, not run on hardware
//
#include <hip/hip_runtime.h>
#include <stdint.h>

// ---------------------------------------------------------------------------
// Problem sizes (from reference)
// ---------------------------------------------------------------------------
#define IN_DIM   128
#define OUT_DIM  32
#define LIFT     512
#define BATCH    32768

#define ROWS_PER_BLOCK 64          // 4 m-tiles of 16
#define BLK_THREADS    256         // 8 wave32

typedef __attribute__((ext_vector_type(16))) __bf16 v16bf;
typedef __attribute__((ext_vector_type(8)))  float  v8f;

union FragU { v16bf v; uint4 q[2]; };
union AccU  { v8f v; float f[8]; };

// ---------------------------------------------------------------------------
// bf16 helpers (round-to-nearest-even)
// ---------------------------------------------------------------------------
__device__ __forceinline__ uint16_t f2bf(float f) {
    uint32_t u = __float_as_uint(f);
    u += 0x7FFFu + ((u >> 16) & 1u);
    return (uint16_t)(u >> 16);
}
__device__ __forceinline__ float bf2f(uint16_t h) {
    return __uint_as_float(((uint32_t)h) << 16);
}

// Branch-free tanh: tanh(x) = sign(x) * (1 - e) / (1 + e),  e = exp(-2|x|).
// Lowers to v_exp_f32 + v_rcp_f32, no EXEC divergence (keeps the WMMA
// pipeline free of cmpx/branch overhead).
__device__ __forceinline__ float fast_tanh(float v) {
    float av = __builtin_fabsf(v);
    float e  = __expf(-2.0f * av);
    float r  = __fdividef(1.0f - e, 1.0f + e);
    return __builtin_copysignf(r, v);
}

// ---------------------------------------------------------------------------
// WMMA fragment loaders.
//
// A fragment (16x32 bf16, M x K): lane m = lane&15, half = lane>>4 holds
//   K = [half*8, half*8+8) in VGPR0..3 and K = [16+half*8, +8) in VGPR4..7
//   -> two contiguous 16-byte loads.
//
// B fragment (32x16 bf16, K x N) with column n K-contiguous in memory
// (B[k][n] = W[n][k], W row-major, leading dim ld): lane n = lane&15 holds
//   K = [half*8, half*8+8) and K = [16+half*8, +8) -> two 16-byte loads.
// ---------------------------------------------------------------------------
__device__ __forceinline__ v16bf load_a_frag(const uint16_t* __restrict__ base,
                                             int ld, int row0, int k0,
                                             int m, int half) {
    const uint16_t* p = base + (size_t)(row0 + m) * ld + k0 + half * 8;
    FragU f;
    f.q[0] = *(const uint4*)(p);
    f.q[1] = *(const uint4*)(p + 16);
    return f.v;
}

__device__ __forceinline__ v16bf load_b_frag(const uint16_t* __restrict__ wbase,
                                             int ld, int n0, int k0,
                                             int n, int half) {
    const uint16_t* p = wbase + (size_t)(n0 + n) * ld + k0 + half * 8;
    FragU f;
    f.q[0] = *(const uint4*)(p);
    f.q[1] = *(const uint4*)(p + 16);
    return f.v;
}

__device__ __forceinline__ v8f wmma_bf16(v16bf a, v16bf b, v8f c) {
    return __builtin_amdgcn_wmma_f32_16x16x32_bf16(false, a, false, b,
                                                   (short)0, c, false, false);
}

// ---------------------------------------------------------------------------
// Precompute: Q = L @ L^T  (bf16, 512x512, row-major) — one thread per entry
// ---------------------------------------------------------------------------
__global__ __launch_bounds__(256)
void build_q_kernel(const float* __restrict__ Lv, uint16_t* __restrict__ Qbf) {
    int idx = blockIdx.x * 256 + threadIdx.x;       // 512*512 entries
    int i = idx >> 9;
    int j = idx & 511;
    const float* Li = Lv + ((size_t)i * (i + 1)) / 2;
    const float* Lj = Lv + ((size_t)j * (j + 1)) / 2;
    int kmax = (i < j) ? i : j;
    float s = 0.f;
    for (int k = 0; k <= kmax; ++k) s += Li[k] * Lj[k];
    Qbf[idx] = f2bf(s);
}

// ---------------------------------------------------------------------------
// Precompute: f32 -> bf16 conversion
// ---------------------------------------------------------------------------
__global__ __launch_bounds__(256)
void cvt_bf16_kernel(const float* __restrict__ src, uint16_t* __restrict__ dst,
                     int n) {
    int i = blockIdx.x * 256 + threadIdx.x;
    if (i < n) dst[i] = f2bf(src[i]);
}

// ---------------------------------------------------------------------------
// Fused kernel:
//   phi = tanh(x @ W1^T + b1)           (kept in LDS as bf16, 64x512)
//   pi  = phi @ W2^T                    -> out[0 .. BATCH*32)
//   y   = sum((phi@Q + qb) * phi) + qc  -> out[BATCH*32 .. BATCH*33)
// ---------------------------------------------------------------------------
__global__ __launch_bounds__(BLK_THREADS)
void fused_koopman_kernel(const uint16_t* __restrict__ xbf,
                          const uint16_t* __restrict__ W1bf,
                          const float* __restrict__ b1,
                          const uint16_t* __restrict__ W2bf,
                          const uint16_t* __restrict__ Qbf,
                          const float* __restrict__ qb,
                          const float* __restrict__ qc,
                          float* __restrict__ out) {
    __shared__ uint16_t phi_s[ROWS_PER_BLOCK * LIFT];   // 64 KB
    __shared__ float    y_s[ROWS_PER_BLOCK];

    const int tid   = threadIdx.x;
    const int lane  = tid & 31;
    const int half  = lane >> 4;
    const int l15   = lane & 15;
    const int wave  = tid >> 5;
    const int mtile = wave & 3;          // 0..3 : 16-row tile within block
    const int nhalf = wave >> 2;         // 0..1 : which half of the 512 cols
    const int row0  = blockIdx.x * ROWS_PER_BLOCK;

    if (tid < ROWS_PER_BLOCK) y_s[tid] = 0.f;

    // ---------------- GEMM1: phi = tanh(x @ W1^T + b1) --------------------
    // A: x rows [row0 + mtile*16, +16), K = 128 (global bf16, ld = 128)
    v16bf a0 = load_a_frag(xbf, IN_DIM, row0 + mtile * 16, 0,  l15, half);
    v16bf a1 = load_a_frag(xbf, IN_DIM, row0 + mtile * 16, 32, l15, half);
    v16bf a2 = load_a_frag(xbf, IN_DIM, row0 + mtile * 16, 64, l15, half);
    v16bf a3 = load_a_frag(xbf, IN_DIM, row0 + mtile * 16, 96, l15, half);

    // Two n-tiles per iteration: two independent accumulator chains keep the
    // matrix pipe busy (no D->C serial dependency between consecutive WMMAs).
    for (int p = 0; p < 8; ++p) {
        const int n0a = (nhalf * 16 + 2 * p) * 16;
        const int n0b = n0a + 16;

        AccU acc0, acc1;
        #pragma unroll
        for (int r = 0; r < 8; ++r) { acc0.f[r] = 0.f; acc1.f[r] = 0.f; }

        #pragma unroll
        for (int kt = 0; kt < 4; ++kt) {
            v16bf b0 = load_b_frag(W1bf, IN_DIM, n0a, kt * 32, l15, half);
            v16bf b1f = load_b_frag(W1bf, IN_DIM, n0b, kt * 32, l15, half);
            v16bf a = (kt == 0) ? a0 : (kt == 1) ? a1 : (kt == 2) ? a2 : a3;
            acc0.v = wmma_bf16(a, b0,  acc0.v);
            acc1.v = wmma_bf16(a, b1f, acc1.v);
        }

        const float biasa = b1[n0a + l15];
        const float biasb = b1[n0b + l15];
        #pragma unroll
        for (int r = 0; r < 8; ++r) {
            const size_t rowoff = (size_t)(mtile * 16 + r + 8 * half) * LIFT + l15;
            phi_s[rowoff + n0a] = f2bf(fast_tanh(acc0.f[r] + biasa));
            phi_s[rowoff + n0b] = f2bf(fast_tanh(acc1.f[r] + biasb));
        }
    }
    __syncthreads();

    // ---------------- GEMM2: t = phi @ Q ; y += sum((t+qb)*phi) -----------
    float yl[8];
    #pragma unroll
    for (int r = 0; r < 8; ++r) yl[r] = 0.f;

    for (int p = 0; p < 8; ++p) {
        const int n0a = (nhalf * 16 + 2 * p) * 16;
        const int n0b = n0a + 16;

        // prefetch upcoming Q rows into L2 (global_prefetch_b8)
        __builtin_prefetch(Qbf + (size_t)(n0b + 16 + l15) * LIFT, 0, 1);

        AccU acc0, acc1;
        #pragma unroll
        for (int r = 0; r < 8; ++r) { acc0.f[r] = 0.f; acc1.f[r] = 0.f; }

        #pragma unroll 4
        for (int kt = 0; kt < 16; ++kt) {
            v16bf a  = load_a_frag(phi_s, LIFT, mtile * 16, kt * 32, l15, half);
            v16bf b0 = load_b_frag(Qbf, LIFT, n0a, kt * 32, l15, half);
            v16bf b1f = load_b_frag(Qbf, LIFT, n0b, kt * 32, l15, half);
            acc0.v = wmma_bf16(a, b0,  acc0.v);
            acc1.v = wmma_bf16(a, b1f, acc1.v);
        }

        const float qba = qb[n0a + l15];
        const float qbb = qb[n0b + l15];
        #pragma unroll
        for (int r = 0; r < 8; ++r) {
            const size_t rowoff = (size_t)(mtile * 16 + r + 8 * half) * LIFT + l15;
            yl[r] += (acc0.f[r] + qba) * bf2f(phi_s[rowoff + n0a]);
            yl[r] += (acc1.f[r] + qbb) * bf2f(phi_s[rowoff + n0b]);
        }
    }

    // reduce each yl[r] across the 16 lanes of this half-wave
    #pragma unroll
    for (int r = 0; r < 8; ++r) {
        float v = yl[r];
        v += __shfl_xor(v, 1, 32);
        v += __shfl_xor(v, 2, 32);
        v += __shfl_xor(v, 4, 32);
        v += __shfl_xor(v, 8, 32);
        if (l15 == 0)
            atomicAdd(&y_s[mtile * 16 + r + 8 * half], v);
    }

    // ---------------- GEMM3: pi = phi @ W2^T ------------------------------
    {
        const int n0 = nhalf * 16;    // N = 32 -> two n-tiles, one per nhalf
        AccU acc;
        #pragma unroll
        for (int r = 0; r < 8; ++r) acc.f[r] = 0.f;

        #pragma unroll 4
        for (int kt = 0; kt < 16; ++kt) {
            v16bf a = load_a_frag(phi_s, LIFT, mtile * 16, kt * 32, l15, half);
            v16bf b = load_b_frag(W2bf,  LIFT, n0,        kt * 32, l15, half);
            acc.v = wmma_bf16(a, b, acc.v);
        }

        #pragma unroll
        for (int r = 0; r < 8; ++r) {
            int grow = row0 + mtile * 16 + r + 8 * half;
            out[(size_t)grow * OUT_DIM + n0 + l15] = acc.f[r];
        }
    }

    __syncthreads();
    if (tid < ROWS_PER_BLOCK) {
        out[(size_t)BATCH * OUT_DIM + row0 + tid] = y_s[tid] + qc[0];
    }
}

// ---------------------------------------------------------------------------
// Launch
// ---------------------------------------------------------------------------
extern "C" void kernel_launch(void* const* d_in, const int* in_sizes, int n_in,
                              void* d_out, int out_size, void* d_ws, size_t ws_size,
                              hipStream_t stream) {
    (void)in_sizes; (void)n_in; (void)out_size; (void)ws_size;

    const float* x     = (const float*)d_in[0];
    const float* W1    = (const float*)d_in[1];
    const float* b1    = (const float*)d_in[2];
    const float* W2    = (const float*)d_in[3];
    const float* Lvars = (const float*)d_in[4];
    const float* qb    = (const float*)d_in[5];
    const float* qc    = (const float*)d_in[6];
    float* out = (float*)d_out;

    // workspace layout (16B-aligned offsets)
    char* ws = (char*)d_ws;
    uint16_t* Qbf  = (uint16_t*)(ws);                           // 512*512*2 = 524288
    uint16_t* W1bf = (uint16_t*)(ws + 524288);                  // 512*128*2 = 131072
    uint16_t* W2bf = (uint16_t*)(ws + 524288 + 131072);         // 32*512*2  =  32768
    uint16_t* xbf  = (uint16_t*)(ws + 524288 + 131072 + 32768); // 32768*128*2 = 8 MB

    // Precompute Q = L L^T (bf16) and bf16 copies of W1, W2, x
    build_q_kernel<<<(LIFT * LIFT) / 256, 256, 0, stream>>>(Lvars, Qbf);
    cvt_bf16_kernel<<<(LIFT * IN_DIM + 255) / 256, 256, 0, stream>>>(W1, W1bf, LIFT * IN_DIM);
    cvt_bf16_kernel<<<(OUT_DIM * LIFT + 255) / 256, 256, 0, stream>>>(W2, W2bf, OUT_DIM * LIFT);
    cvt_bf16_kernel<<<(BATCH * IN_DIM + 255) / 256, 256, 0, stream>>>(x, xbf, BATCH * IN_DIM);

    // Fused: phi -> (pi, y)
    fused_koopman_kernel<<<BATCH / ROWS_PER_BLOCK, BLK_THREADS, 0, stream>>>(
        xbf, W1bf, b1, W2bf, Qbf, qb, qc, out);
}
